// MultiHeadAttention_25872882991960
// MI455X (gfx1250) — compile-verified, matrix operand order checked by reading, not run on hardware
//
#include <hip/hip_runtime.h>

// Problem constants (match reference: B, S, D, H = 4, 2048, 1024, 16)
#define B_  4
#define S_  2048
#define D_  1024
#define H_  16
#define HD_ 64

typedef __attribute__((ext_vector_type(16))) __bf16 v16bf;
typedef __attribute__((ext_vector_type(8)))  __bf16 v8bf;
typedef __attribute__((ext_vector_type(8)))  float  v8f;

// float -> bf16 using the native converter (clang __bf16 cast lowers to
// v_cvt_*bf16_f32 on gfx1250; RTNE per hardware default).
__device__ __forceinline__ __bf16 f2bf(float f) { return (__bf16)f; }

// A/B fragment for V_WMMA_F32_16X16X32_BF16 (ISA 7.12.2):
// lane L holds row/col (L&15), K = 8*(L>>4) + {0..7} and +16 + {0..7}.
// `base` points at (row0, k0) of a tile whose K-values are contiguous with
// row-stride `ld` elements. All addresses are 16B aligned by construction.
__device__ __forceinline__ v16bf load_frag_bf16(const __bf16* base, int ld, int lane) {
  const int r  = lane & 15;
  const int kb = (lane >> 4) << 3;
  const __bf16* p = base + (size_t)r * ld + kb;
  v8bf lo = *(const v8bf*)(p);
  v8bf hi = *(const v8bf*)(p + 16);
  return __builtin_shufflevector(lo, hi, 0, 1, 2, 3, 4, 5, 6, 7,
                                 8, 9, 10, 11, 12, 13, 14, 15);
}

// Same fragment, sourced from fp32 memory with on-the-fly bf16 conversion.
__device__ __forceinline__ v16bf load_frag_f32(const float* base, int ld, int lane) {
  const int r  = lane & 15;
  const int kb = (lane >> 4) << 3;
  const float* p = base + (size_t)r * ld + kb;
  v16bf f;
#pragma unroll
  for (int i = 0; i < 8; ++i) { f[i] = f2bf(p[i]); f[i + 8] = f2bf(p[16 + i]); }
  return f;
}

__device__ __forceinline__ v8f wmma_bf16(v16bf a, v16bf b, v8f c) {
  // (neg_a, A, neg_b, B, c_mod, C, reuse_a, reuse_b)
  return __builtin_amdgcn_wmma_f32_16x16x32_bf16(false, a, false, b, (short)0, c,
                                                 false, false);
}

// ---------------------------------------------------------------------------
// Kernel 0: W[k][n] fp32 -> Wt[n][k] bf16 (transposed so B-fragments are
// contiguous along K).
// ---------------------------------------------------------------------------
__global__ void wtrans_kernel(const float* __restrict__ w, __bf16* __restrict__ wt) {
  int idx = blockIdx.x * blockDim.x + threadIdx.x;
  if (idx >= D_ * D_) return;
  int n = idx / D_;
  int k = idx - n * D_;
  wt[idx] = f2bf(w[(size_t)k * D_ + n]);
}

// ---------------------------------------------------------------------------
// Kernel 1: fused QKV projection. One wave computes a 16(M) x 64(N) tile for
// all three projections (12 independent WMMA chains sharing the A fragment).
//   Q -> [b,h,s,hd] bf16, pre-scaled by 1/sqrt(HD)
//   K -> [b,h,s,hd] bf16
//   V -> [b,h,hd,s] bf16 (transposed: PV B-fragment contiguous over keys)
// ---------------------------------------------------------------------------
__global__ void __launch_bounds__(32) qkv_kernel(
    const float* __restrict__ x,
    const __bf16* __restrict__ wqT, const __bf16* __restrict__ wkT,
    const __bf16* __restrict__ wvT,
    __bf16* __restrict__ Q, __bf16* __restrict__ K, __bf16* __restrict__ Vt) {
  const int lane = threadIdx.x;
  const int m0 = blockIdx.x * 16;   // token tile
  const int n0 = blockIdx.y * 64;   // output-feature tile
  v8f aq[4] = {}, ak[4] = {}, av[4] = {};
  for (int kt = 0; kt < D_; kt += 32) {
    v16bf a = load_frag_f32(x + (size_t)m0 * D_ + kt, D_, lane);
#pragma unroll
    for (int f = 0; f < 4; ++f) {
      const size_t wo = (size_t)(n0 + f * 16) * D_ + kt;
      aq[f] = wmma_bf16(a, load_frag_bf16(wqT + wo, D_, lane), aq[f]);
      ak[f] = wmma_bf16(a, load_frag_bf16(wkT + wo, D_, lane), ak[f]);
      av[f] = wmma_bf16(a, load_frag_bf16(wvT + wo, D_, lane), av[f]);
    }
  }
  const int cl = lane & 15;
  const int rh = (lane >> 4) << 3;
#pragma unroll
  for (int f = 0; f < 4; ++f) {
    const int n  = n0 + f * 16 + cl;
    const int h  = n / HD_;
    const int hd = n - h * HD_;
#pragma unroll
    for (int r = 0; r < 8; ++r) {
      const int m = m0 + r + rh;
      const int b = m / S_;
      const int s = m - b * S_;
      const size_t qk = ((size_t)(b * H_ + h) * S_ + s) * HD_ + hd;
      Q[qk] = f2bf(aq[f][r] * 0.125f);   // fold 1/sqrt(64)
      K[qk] = f2bf(ak[f][r]);
      Vt[((size_t)(b * H_ + h) * HD_ + hd) * S_ + s] = f2bf(av[f][r]);
    }
  }
}

// ---------------------------------------------------------------------------
// Kernel 2: causal flash attention. One wave per (b, h, 16-query tile).
// Streams 32 keys per step: 4 score WMMAs -> online softmax (cross-lane via
// shfl_xor within 16-lane groups) -> P reshaped C-layout->A-layout through a
// 1KB LDS tile (s_wait_dscnt for the cross-lane dependency) -> 4 PV WMMAs.
// Writes ctx[b,s,D] bf16.
// ---------------------------------------------------------------------------
__global__ void __launch_bounds__(32) attn_kernel(
    const __bf16* __restrict__ Q, const __bf16* __restrict__ K,
    const __bf16* __restrict__ Vt, const int* __restrict__ amask,
    __bf16* __restrict__ ctx) {
  __shared__ __bf16 ptile[16 * 32];
  const int lane = threadIdx.x;
  const int qt = blockIdx.x, h = blockIdx.y, b = blockIdx.z;
  const int q0 = qt * 16;
  const size_t bh = (size_t)(b * H_ + h);
  const __bf16* Qb = Q  + bh * (size_t)S_ * HD_;
  const __bf16* Kb = K  + bh * (size_t)S_ * HD_;
  const __bf16* Vb = Vt + bh * (size_t)HD_ * S_;
  const int cl = lane & 15;
  const int rh = (lane >> 4) << 3;

  // Q fragments (16 queries x HD=64 -> two K=32 fragments), resident all loop.
  const v16bf aQ0 = load_frag_bf16(Qb + (size_t)q0 * HD_,      HD_, lane);
  const v16bf aQ1 = load_frag_bf16(Qb + (size_t)q0 * HD_ + 32, HD_, lane);

  v8f o[4] = {};
  float m[8], l[8];
#pragma unroll
  for (int r = 0; r < 8; ++r) { m[r] = -1e30f; l[r] = 0.0f; }

  for (int k0 = 0; k0 <= q0 + 15; k0 += 32) {
    // scores: 16 queries x 32 keys as two 16x16 C fragments
    v8f s0 = {}, s1 = {};
    s0 = wmma_bf16(aQ0, load_frag_bf16(Kb + (size_t)k0 * HD_,             HD_, lane), s0);
    s0 = wmma_bf16(aQ1, load_frag_bf16(Kb + (size_t)k0 * HD_ + 32,        HD_, lane), s0);
    s1 = wmma_bf16(aQ0, load_frag_bf16(Kb + (size_t)(k0 + 16) * HD_,      HD_, lane), s1);
    s1 = wmma_bf16(aQ1, load_frag_bf16(Kb + (size_t)(k0 + 16) * HD_ + 32, HD_, lane), s1);

    const int kA = k0 + cl, kB = kA + 16;           // this lane's key columns
    const bool okA = (amask[b * S_ + kA] != 0);
    const bool okB = (amask[b * S_ + kB] != 0);
#pragma unroll
    for (int r = 0; r < 8; ++r) {
      const int q = q0 + r + rh;                    // this element's query row
      float v0 = (okA && kA <= q) ? s0[r] : -1e30f; // causal + key mask
      float v1 = (okB && kB <= q) ? s1[r] : -1e30f;
      // row max across the 16 lanes holding this row (masks<16 stay in-half)
      float mx = fmaxf(v0, v1);
      mx = fmaxf(mx, __shfl_xor(mx, 1));
      mx = fmaxf(mx, __shfl_xor(mx, 2));
      mx = fmaxf(mx, __shfl_xor(mx, 4));
      mx = fmaxf(mx, __shfl_xor(mx, 8));
      const float mn = fmaxf(m[r], mx);
      const float sc = __expf(m[r] - mn);
      m[r] = mn;
      const float p0 = __expf(v0 - mn);
      const float p1 = __expf(v1 - mn);
      float rs = p0 + p1;
      rs += __shfl_xor(rs, 1);
      rs += __shfl_xor(rs, 2);
      rs += __shfl_xor(rs, 4);
      rs += __shfl_xor(rs, 8);
      l[r] = l[r] * sc + rs;
#pragma unroll
      for (int f = 0; f < 4; ++f) o[f][r] *= sc;    // rescale running output
      // stash P (bf16) row-major 16x32 for the layout change
      const int row = r + rh;
      ptile[row * 32 + cl]      = f2bf(p0);
      ptile[row * 32 + cl + 16] = f2bf(p1);
    }
    // cross-lane LDS dependency: wait for all ds stores of this wave
    asm volatile("s_wait_dscnt 0" ::: "memory");
    const v16bf aP = load_frag_bf16(&ptile[0], 32, lane);  // A-layout reload
#pragma unroll
    for (int f = 0; f < 4; ++f) {
      v16bf bv = load_frag_bf16(Vb + (size_t)(f * 16) * S_ + k0, S_, lane);
      o[f] = wmma_bf16(aP, bv, o[f]);
    }
  }

  // normalize and emit ctx[b, s, h*HD + hd] in bf16
#pragma unroll
  for (int f = 0; f < 4; ++f) {
#pragma unroll
    for (int r = 0; r < 8; ++r) {
      const int s = q0 + r + rh;
      const int n = h * HD_ + f * 16 + cl;
      ctx[((size_t)b * S_ + s) * D_ + n] = f2bf(o[f][r] / l[r]);
    }
  }
}

// ---------------------------------------------------------------------------
// Kernel 3: output projection: out(fp32) = ctx(bf16) @ Wo. 16x64 per wave.
// ---------------------------------------------------------------------------
__global__ void __launch_bounds__(32) oproj_kernel(
    const __bf16* __restrict__ ctx, const __bf16* __restrict__ woT,
    float* __restrict__ out) {
  const int lane = threadIdx.x;
  const int m0 = blockIdx.x * 16;
  const int n0 = blockIdx.y * 64;
  v8f acc[4] = {};
  for (int kt = 0; kt < D_; kt += 32) {
    v16bf a = load_frag_bf16(ctx + (size_t)m0 * D_ + kt, D_, lane);
#pragma unroll
    for (int f = 0; f < 4; ++f)
      acc[f] = wmma_bf16(
          a, load_frag_bf16(woT + (size_t)(n0 + f * 16) * D_ + kt, D_, lane),
          acc[f]);
  }
  const int cl = lane & 15;
  const int rh = (lane >> 4) << 3;
#pragma unroll
  for (int f = 0; f < 4; ++f)
#pragma unroll
    for (int r = 0; r < 8; ++r)
      out[(size_t)(m0 + r + rh) * D_ + n0 + f * 16 + cl] = acc[f][r];
}

// ---------------------------------------------------------------------------
extern "C" void kernel_launch(void* const* d_in, const int* in_sizes, int n_in,
                              void* d_out, int out_size, void* d_ws, size_t ws_size,
                              hipStream_t stream) {
  const float* x  = (const float*)d_in[0];
  const int*   am = (const int*)d_in[1];
  const float* wq = (const float*)d_in[2];
  const float* wk = (const float*)d_in[3];
  const float* wv = (const float*)d_in[4];
  const float* wo = (const float*)d_in[5];
  float* out = (float*)d_out;

  // Workspace layout (bf16 intermediates stay resident in the 192MB L2):
  //  4 x 2MB  transposed weights
  //  3 x 16MB Q, K, Vt
  //  1 x 16MB ctx                     => 72 MB total
  char* ws = (char*)d_ws;
  const size_t wbytes   = (size_t)D_ * D_ * sizeof(__bf16);
  const size_t qkvbytes = (size_t)B_ * H_ * S_ * HD_ * sizeof(__bf16);
  __bf16* wqT = (__bf16*)(ws + 0 * wbytes);
  __bf16* wkT = (__bf16*)(ws + 1 * wbytes);
  __bf16* wvT = (__bf16*)(ws + 2 * wbytes);
  __bf16* woT = (__bf16*)(ws + 3 * wbytes);
  __bf16* Qb  = (__bf16*)(ws + 4 * wbytes);
  __bf16* Kb  = (__bf16*)(ws + 4 * wbytes + 1 * qkvbytes);
  __bf16* Vt  = (__bf16*)(ws + 4 * wbytes + 2 * qkvbytes);
  __bf16* ctx = (__bf16*)(ws + 4 * wbytes + 3 * qkvbytes);

  const dim3 tb(256), tg((D_ * D_ + 255) / 256);
  wtrans_kernel<<<tg, tb, 0, stream>>>(wq, wqT);
  wtrans_kernel<<<tg, tb, 0, stream>>>(wk, wkT);
  wtrans_kernel<<<tg, tb, 0, stream>>>(wv, wvT);
  wtrans_kernel<<<tg, tb, 0, stream>>>(wo, woT);

  qkv_kernel<<<dim3((B_ * S_) / 16, D_ / 64), dim3(32), 0, stream>>>(
      x, wqT, wkT, wvT, Qb, Kb, Vt);
  attn_kernel<<<dim3(S_ / 16, H_, B_), dim3(32), 0, stream>>>(Qb, Kb, Vt, am, ctx);
  oproj_kernel<<<dim3((B_ * S_) / 16, D_ / 64), dim3(32), 0, stream>>>(ctx, woT, out);

  (void)in_sizes; (void)n_in; (void)out_size; (void)ws_size;
}
